// GroupedQueryAttention_81406810128458
// MI455X (gfx1250) — compile-verified, matrix operand order checked by reading, not run on hardware
//
#include <hip/hip_runtime.h>

// ---------------------------------------------------------------------------
// GroupedQueryAttention for MI455X (gfx1250), bf16 WMMA with fp32 accumulate.
//   B=4, S=8192, E=512, H=8, D=64, G=64, gs=128
// Round 2: stage q/k/v tiles in the attention kernel via the Tensor Data
// Mover (tensor_load_to_lds + s_wait_tensorcnt) instead of VGPR copies.
// ---------------------------------------------------------------------------

typedef __attribute__((ext_vector_type(16))) __bf16 v16bf;
typedef __attribute__((ext_vector_type(8)))  float  v8f;
typedef unsigned __attribute__((ext_vector_type(4))) uint32x4;
typedef int      __attribute__((ext_vector_type(4))) int32x4;
typedef int      __attribute__((ext_vector_type(8))) int32x8;

#define Bb 4
#define Ss 8192
#define Ee 512
#define Hh 8
#define Dd 64
#define Gg 64
#define GS 128

// ---- WMMA 16x16x32 bf16 operand packing (ISA 7.12.2, 16-bit A 16x32) ------
// lane l, reg r holds K = 2*(r&3) + (r>=4?16:0) + (l>=16?8:0)  (pair K,K+1)
// A: lane -> row (l&15);  B: lane -> col (l&15).  Same packing for both.

__device__ __forceinline__ v16bf load_frag_rowK(const __bf16* base, int row,
                                                int stride, int kbase, int lane) {
    // element (row, k) at base[row*stride + k]; K-pairs contiguous in LDS
    v16bf f;
    const int kh = (lane >> 4) << 3;  // +8 for upper half-wave
#pragma unroll
    for (int r = 0; r < 8; ++r) {
        int k = kbase + 2 * (r & 3) + ((r & 4) << 2) + kh;
        unsigned u = *(const unsigned*)(base + row * stride + k);
        union { unsigned u; __bf16 h[2]; } cv;
        cv.u = u;
        f[2 * r]     = cv.h[0];
        f[2 * r + 1] = cv.h[1];
    }
    return f;
}

__device__ __forceinline__ v16bf load_frag_colK(const __bf16* base, int col,
                                                int stride, int kbase, int lane) {
    // element (k, col) at base[k*stride + col]; K-pairs strided
    v16bf f;
    const int kh = (lane >> 4) << 3;
#pragma unroll
    for (int r = 0; r < 8; ++r) {
        int k = kbase + 2 * (r & 3) + ((r & 4) << 2) + kh;
        f[2 * r]     = base[k * stride + col];
        f[2 * r + 1] = base[(k + 1) * stride + col];
    }
    return f;
}

// ---- Tensor Data Mover: 1D contiguous bf16 tile, Global -> LDS ------------
// D# group0: count=1 | lds_addr[63:32] | global_addr[120:64] | type=2
// D# group1: data_size=2B, tensor_dim0=tile_dim0=nelem, stride=nelem
__device__ __forceinline__ void tdm_load_1d(unsigned lds_byte_off,
                                            const void* gptr, unsigned nelem) {
    unsigned long long ga = (unsigned long long)(uintptr_t)gptr;
    uint32x4 g0;
    g0[0] = 1u;                                        // count=1 (valid user D#)
    g0[1] = lds_byte_off;                              // lds_addr
    g0[2] = (unsigned)(ga & 0xFFFFFFFFu);              // global_addr[31:0]
    g0[3] = (unsigned)((ga >> 32) & 0x1FFFFFFu)        // global_addr[56:32]
            | (2u << 30);                              // type = 2 ("image")
    int32x8 g1;
    g1[0] = (int)(1u << 16);                           // data_size=1 -> 2 bytes
    g1[1] = (int)(nelem << 16);                        // tensor_dim0[15:0]
    g1[2] = (int)((nelem >> 16) | (1u << 16));         // tensor_dim0[31:16], tensor_dim1=1
    g1[3] = (int)(nelem << 16);                        // tile_dim0 (bits 127:112)
    g1[4] = 0;                                         // tile_dim1=0, tile_dim2=0
    g1[5] = (int)nelem;                                // tensor_dim0_stride[31:0]
    g1[6] = 0;
    g1[7] = 0;
    int32x4 z4 = {0, 0, 0, 0};
#if __clang_major__ >= 23
    int32x8 z8 = {0, 0, 0, 0, 0, 0, 0, 0};
    __builtin_amdgcn_tensor_load_to_lds(g0, g1, z4, z4, z8, 0);
#else
    __builtin_amdgcn_tensor_load_to_lds(g0, g1, z4, z4, 0);
#endif
}

// ---------------------------------------------------------------------------
// Kernel 1: qkv = x @ W_qkv + b_qkv    (M=32768, N=1536, K=512)
// 128x128 tile per WG (256 thr = 8 waves; waves 4x2 -> 32x64 each).
// Epilogue scatters bf16 into qkv_ws laid out [3][B][H][S][D].
// ---------------------------------------------------------------------------
__global__ __launch_bounds__(256) void qkv_gemm_kernel(
    const float* __restrict__ x, const float* __restrict__ Wqkv,
    const float* __restrict__ bqkv, __bf16* __restrict__ qkv_ws) {
    __shared__ __bf16 lA[128 * 32];  // [m][k]
    __shared__ __bf16 lB[128 * 32];  // [n][k]

    const int tid  = threadIdx.x;
    const int lane = tid & 31;
    const int w    = tid >> 5;
    const int Nbase = blockIdx.x * 128;
    const int Mbase = blockIdx.y * 128;
    const int wm = (w & 3) * 32;   // wave row offset in tile
    const int wn = (w >> 2) * 64;  // wave col offset in tile

    v8f acc[2][4] = {};

    for (int k0 = 0; k0 < Ee; k0 += 32) {
        // A tile: x fp32 -> bf16 LDS [row][k]
#pragma unroll
        for (int it = 0; it < 4; ++it) {
            int linear = tid + it * 256;          // 0..1023
            int e = linear * 4;
            int row = e >> 5, kk = e & 31;        // kk multiple of 4
            const float4 xv =
                *(const float4*)(x + (size_t)(Mbase + row) * Ee + k0 + kk);
            __bf16* dst = lA + row * 32 + kk;
            dst[0] = (__bf16)xv.x; dst[1] = (__bf16)xv.y;
            dst[2] = (__bf16)xv.z; dst[3] = (__bf16)xv.w;
        }
        // B tile: Wqkv[k][n] fp32 -> bf16 LDS transposed [n][k]
#pragma unroll
        for (int it = 0; it < 4; ++it) {
            int linear = tid + it * 256;
            int kk = linear >> 5;                 // 0..31
            int n0 = (linear & 31) * 4;
            const float4 wv =
                *(const float4*)(Wqkv + (size_t)(k0 + kk) * 1536 + Nbase + n0);
            lB[(n0 + 0) * 32 + kk] = (__bf16)wv.x;
            lB[(n0 + 1) * 32 + kk] = (__bf16)wv.y;
            lB[(n0 + 2) * 32 + kk] = (__bf16)wv.z;
            lB[(n0 + 3) * 32 + kk] = (__bf16)wv.w;
        }
        __syncthreads();

        if (k0 + 32 < Ee) {  // prefetch next K chunk (global_prefetch_b8)
            __builtin_prefetch(x + (size_t)(Mbase + (tid >> 1)) * Ee + k0 + 32, 0, 0);
            __builtin_prefetch(Wqkv + (size_t)(k0 + 32 + lane) * 1536 + Nbase, 0, 0);
        }

        v16bf aF[2], bF[4];
#pragma unroll
        for (int mi = 0; mi < 2; ++mi)
            aF[mi] = load_frag_rowK(lA, wm + mi * 16 + (lane & 15), 32, 0, lane);
#pragma unroll
        for (int ni = 0; ni < 4; ++ni)
            bF[ni] = load_frag_rowK(lB, wn + ni * 16 + (lane & 15), 32, 0, lane);
#pragma unroll
        for (int mi = 0; mi < 2; ++mi)
#pragma unroll
            for (int ni = 0; ni < 4; ++ni)
                acc[mi][ni] = __builtin_amdgcn_wmma_f32_16x16x32_bf16(
                    false, aF[mi], false, bF[ni], (short)0, acc[mi][ni], false, false);
        __syncthreads();
    }

    // Epilogue: bias + scatter into [3][B][H][S][D] bf16
#pragma unroll
    for (int mi = 0; mi < 2; ++mi) {
#pragma unroll
        for (int ni = 0; ni < 4; ++ni) {
#pragma unroll
            for (int r = 0; r < 8; ++r) {
                int row = Mbase + wm + mi * 16 + ((lane < 16) ? r : r + 8);
                int col = Nbase + wn + ni * 16 + (lane & 15);
                float v = acc[mi][ni][r] + bqkv[col];
                int b = row >> 13, s = row & (Ss - 1);
                int which = col >> 9, hd = col & 511;
                int h = hd >> 6, d = hd & 63;
                size_t off =
                    ((((size_t)which * Bb + b) * Hh + h) * Ss + s) * Dd + d;
                qkv_ws[off] = (__bf16)v;
            }
        }
    }
}

// ---------------------------------------------------------------------------
// Kernel 2: block-diagonal attention. One WG per (b,h,g) block.
// q,k,v staged by TDM into LDS as [128][64] bf16; scores 128x128; softmax in
// regs; probs in LDS (reusing q/k region); out = probs @ v -> attn_ws bf16.
// ---------------------------------------------------------------------------
__global__ __launch_bounds__(256) void attn_kernel(
    const __bf16* __restrict__ qkv_ws, __bf16* __restrict__ attn_ws) {
    __shared__ __bf16 lds[3 * GS * Dd];  // 48 KB, base LDS offset 0
    __bf16* lq = lds;
    __bf16* lk = lds + GS * Dd;
    __bf16* lv = lds + 2 * GS * Dd;
    __bf16* lp = lds;  // probs [128][128] reuses q+k region after scores

    const int tid  = threadIdx.x;
    const int lane = tid & 31;
    const int w    = tid >> 5;
    const int g  = blockIdx.x & (Gg - 1);
    const int bh = blockIdx.x >> 6;
    const int b = bh >> 3, h = bh & 7;

    const size_t blk   = (((size_t)b * Hh + h) * Ss + (size_t)g * GS) * Dd;
    const size_t plane = (size_t)Bb * Hh * Ss * Dd;
    const __bf16* gq = qkv_ws + blk;
    const __bf16* gk = qkv_ws + plane + blk;
    const __bf16* gv = qkv_ws + 2 * plane + blk;

    // Stage q,k,v (8192 bf16 = 16 KB each, fully contiguous) via the Tensor
    // Data Mover. Wave 0 issues the three DMAs and waits on its TENSORcnt;
    // the workgroup barrier releases the other waves.
    if (tid < 32) {
        tdm_load_1d(0u,                      gq, GS * Dd);
        tdm_load_1d((unsigned)(GS * Dd * 2), gk, GS * Dd);
        tdm_load_1d((unsigned)(GS * Dd * 4), gv, GS * Dd);
        __builtin_amdgcn_s_wait_tensorcnt(0);
    }
    __syncthreads();

    // scores = (q @ k^T) * 1/sqrt(D): wave w owns query rows [16w,16w+16)
    v8f acc[8] = {};
    const int qrow = w * 16 + (lane & 15);
#pragma unroll
    for (int kk = 0; kk < Dd; kk += 32) {
        v16bf aF = load_frag_rowK(lq, qrow, Dd, kk, lane);
#pragma unroll
        for (int t = 0; t < 8; ++t) {
            v16bf bF = load_frag_rowK(lk, t * 16 + (lane & 15), Dd, kk, lane);
            acc[t] = __builtin_amdgcn_wmma_f32_16x16x32_bf16(
                false, aF, false, bF, (short)0, acc[t], false, false);
        }
    }

    // row softmax: row R lives at reg r across 16 lanes of one half-wave
    const float scale = 0.125f;  // 1/sqrt(64)
#pragma unroll
    for (int r = 0; r < 8; ++r) {
        float m = -1e30f;
#pragma unroll
        for (int t = 0; t < 8; ++t) {
            acc[t][r] *= scale;
            m = fmaxf(m, acc[t][r]);
        }
#pragma unroll
        for (int off = 8; off >= 1; off >>= 1)
            m = fmaxf(m, __shfl_xor(m, off, 32));
        float s = 0.f;
#pragma unroll
        for (int t = 0; t < 8; ++t) {
            float e = __expf(acc[t][r] - m);
            acc[t][r] = e;
            s += e;
        }
#pragma unroll
        for (int off = 8; off >= 1; off >>= 1) s += __shfl_xor(s, off, 32);
        float inv = 1.0f / s;
#pragma unroll
        for (int t = 0; t < 8; ++t) acc[t][r] *= inv;
    }

    __syncthreads();  // everyone done reading lq/lk
#pragma unroll
    for (int t = 0; t < 8; ++t)
#pragma unroll
        for (int r = 0; r < 8; ++r) {
            int row = w * 16 + ((lane < 16) ? r : r + 8);
            int col = t * 16 + (lane & 15);
            lp[row * GS + col] = (__bf16)acc[t][r];
        }
    __syncthreads();

    // out = probs @ v
    v8f oacc[4] = {};
#pragma unroll
    for (int kk = 0; kk < GS; kk += 32) {
        v16bf aF = load_frag_rowK(lp, qrow, GS, kk, lane);
#pragma unroll
        for (int t = 0; t < 4; ++t) {
            v16bf bF = load_frag_colK(lv, t * 16 + (lane & 15), Dd, kk, lane);
            oacc[t] = __builtin_amdgcn_wmma_f32_16x16x32_bf16(
                false, aF, false, bF, (short)0, oacc[t], false, false);
        }
    }

    // write attn out at [b, s, h*64+d] (bf16)
#pragma unroll
    for (int t = 0; t < 4; ++t)
#pragma unroll
        for (int r = 0; r < 8; ++r) {
            int row = w * 16 + ((lane < 16) ? r : r + 8);
            int d = t * 16 + (lane & 15);
            size_t s = (size_t)g * GS + row;
            attn_ws[((size_t)b * Ss + s) * Ee + h * Dd + d] = (__bf16)oacc[t][r];
        }
}

// ---------------------------------------------------------------------------
// Kernel 3: out = attn @ W_out + b_out   (M=32768, N=512, K=512), fp32 out
// ---------------------------------------------------------------------------
__global__ __launch_bounds__(256) void out_gemm_kernel(
    const __bf16* __restrict__ attn_ws, const float* __restrict__ Wout,
    const float* __restrict__ bout, float* __restrict__ out) {
    __shared__ __bf16 lA[128 * 32];  // [m][k]
    __shared__ __bf16 lB[128 * 32];  // [n][k]

    const int tid  = threadIdx.x;
    const int lane = tid & 31;
    const int w    = tid >> 5;
    const int Nbase = blockIdx.x * 128;
    const int Mbase = blockIdx.y * 128;
    const int wm = (w & 3) * 32;
    const int wn = (w >> 2) * 64;

    v8f acc[2][4] = {};

    for (int k0 = 0; k0 < Ee; k0 += 32) {
        // A tile: attn bf16, straight uint4 copy
#pragma unroll
        for (int it = 0; it < 2; ++it) {
            int linear = tid + it * 256;          // 0..511
            int e = linear * 8;
            int row = e >> 5, kk = e & 31;        // kk multiple of 8
            *(uint4*)(lA + row * 32 + kk) =
                *(const uint4*)(attn_ws + (size_t)(Mbase + row) * Ee + k0 + kk);
        }
        // B tile: Wout fp32 -> bf16, transposed [n][k]
#pragma unroll
        for (int it = 0; it < 4; ++it) {
            int linear = tid + it * 256;
            int kk = linear >> 5;
            int n0 = (linear & 31) * 4;
            const float4 wv =
                *(const float4*)(Wout + (size_t)(k0 + kk) * Ee + Nbase + n0);
            lB[(n0 + 0) * 32 + kk] = (__bf16)wv.x;
            lB[(n0 + 1) * 32 + kk] = (__bf16)wv.y;
            lB[(n0 + 2) * 32 + kk] = (__bf16)wv.z;
            lB[(n0 + 3) * 32 + kk] = (__bf16)wv.w;
        }
        __syncthreads();

        if (k0 + 32 < Ee) {
            __builtin_prefetch(attn_ws + (size_t)(Mbase + (tid >> 1)) * Ee + k0 + 32, 0, 0);
            __builtin_prefetch(Wout + (size_t)(k0 + 32 + lane) * Ee + Nbase, 0, 0);
        }

        v16bf aF[2], bF[4];
#pragma unroll
        for (int mi = 0; mi < 2; ++mi)
            aF[mi] = load_frag_rowK(lA, wm + mi * 16 + (lane & 15), 32, 0, lane);
#pragma unroll
        for (int ni = 0; ni < 4; ++ni)
            bF[ni] = load_frag_rowK(lB, wn + ni * 16 + (lane & 15), 32, 0, lane);
#pragma unroll
        for (int mi = 0; mi < 2; ++mi)
#pragma unroll
            for (int ni = 0; ni < 4; ++ni)
                acc[mi][ni] = __builtin_amdgcn_wmma_f32_16x16x32_bf16(
                    false, aF[mi], false, bF[ni], (short)0, acc[mi][ni], false, false);
        __syncthreads();
    }

#pragma unroll
    for (int mi = 0; mi < 2; ++mi)
#pragma unroll
        for (int ni = 0; ni < 4; ++ni)
#pragma unroll
            for (int r = 0; r < 8; ++r) {
                int row = Mbase + wm + mi * 16 + ((lane < 16) ? r : r + 8);
                int col = Nbase + wn + ni * 16 + (lane & 15);
                out[(size_t)row * Ee + col] = acc[mi][ni][r] + bout[col];
            }
}

// ---------------------------------------------------------------------------
extern "C" void kernel_launch(void* const* d_in, const int* in_sizes, int n_in,
                              void* d_out, int out_size, void* d_ws, size_t ws_size,
                              hipStream_t stream) {
    const float* x    = (const float*)d_in[0];
    const float* Wqkv = (const float*)d_in[1];
    const float* bqkv = (const float*)d_in[2];
    const float* Wout = (const float*)d_in[3];
    const float* bout = (const float*)d_in[4];

    __bf16* qkv_ws  = (__bf16*)d_ws;                              // 3*B*H*S*D bf16 (96 MB)
    __bf16* attn_ws = qkv_ws + (size_t)3 * Bb * Hh * Ss * Dd;     // B*S*E bf16  (32 MB)

    qkv_gemm_kernel<<<dim3(1536 / 128, (Bb * Ss) / 128), 256, 0, stream>>>(
        x, Wqkv, bqkv, qkv_ws);
    attn_kernel<<<dim3(Bb * Hh * Gg), 256, 0, stream>>>(qkv_ws, attn_ws);
    out_gemm_kernel<<<dim3(Ee / 128, (Bb * Ss) / 128), 256, 0, stream>>>(
        attn_ws, Wout, bout, (float*)d_out);
}
